// TransportPlanProduct_66185446031909
// MI455X (gfx1250) — matrix operand order repaired and checked by reading, not compiled, not run on hardware
//
#include <hip/hip_runtime.h>

// ---------------------------------------------------------------------------
// TransportPlanProduct for MI455X (gfx1250, wave32, WMMA 16x16x32 f16)
//
// Shapes: B=4 H=8 HW=2048 C=64.  Output = concat(out[B,H,HW,C], attn[B,H,HW,HW])
//
// Roofline: the mandatory attn write (537MB) dominates -> write attn exactly
// once, already normalized, with NON-TEMPORAL stores (TH=NT) so the write-once
// stream doesn't evict the reusable K/V panels from the 192MB L2.  QK tiles
// are recomputed 3x (max / sum / emit) in f16 WMMA; cheap because the
// per-(b,h) K panel (256KB f16) is staged ONCE per workgroup into the WGP's
// 320KB LDS via GLOBAL_LOAD_ASYNC_TO_LDS_B128 (ASYNCcnt + s_wait_asynccnt),
// then all 8 waves x 3 passes read LDS only.  PV consumes normalized P
// directly so `out` needs no post-scale.
// ---------------------------------------------------------------------------

typedef _Float16 v16h __attribute__((ext_vector_type(16)));
typedef _Float16 v8h  __attribute__((ext_vector_type(8)));
typedef float    v8f  __attribute__((ext_vector_type(8)));

#define NB  4
#define NH  8
#define NHW 2048
#define NC  64
#define NBH (NB * NH)                 // 32
#define NROWS (NBH * NHW)             // 65536
#define NELEM ((long)NROWS * NC)      // 4,194,304 per tensor

#define LDSP_BYTES  (8 * 16 * 32 * 2)        // 8KB  P staging (per-wave slices)
#define LDSK_BYTES  (NHW * NC * 2)           // 256KB K panel
#define SMEM_BYTES  (LDSP_BYTES + LDSK_BYTES)

__device__ __forceinline__ v8f wmma_f16(v16h a, v16h b, v8f c) {
  // D = A(16x32 f16) * B(32x16 f16) + C(16x16 f32)
  return __builtin_amdgcn_wmma_f32_16x16x32_f16(
      /*neg_a=*/false, a, /*neg_b=*/false, b,
      /*c_mod=*/(short)0, c, /*reuse_a=*/false, /*reuse_b=*/false);
}

// gfx1250 async global->LDS gather: each lane moves 16B, tracked by ASYNCcnt.
__device__ __forceinline__ void async_g2l_b128(unsigned lds_byte_off,
                                               unsigned long long gaddr) {
  asm volatile("global_load_async_to_lds_b128 %0, %1, off"
               :: "v"(lds_byte_off), "v"(gaddr)
               : "memory");
}
__device__ __forceinline__ void wait_asynccnt0() {
  asm volatile("s_wait_asynccnt 0" ::: "memory");
}

// ---------------------------------------------------------------------------
// Kernel 1: elementwise prep.
//   k_l2 = k/(1e-6+|k|_C), v_l2 = v/(1e-6+|v|_C)
//   Km   = exp(-(1-clip(k_l2*v_l2,0))/0.5)
//   a    = a_rand/(1e-6+|a_rand|_H)          (norm over heads dim)
//   ku   = k * (v/Km) * (k/Km) * a  = k^2 v a / Km^2
// Emits f16: qh = q/8, kuh = ku, vhT = v transposed to [bh][c][hw].
// One wave per (b,h,hw) row; lane l owns c=l and c=l+32.
// ---------------------------------------------------------------------------
__global__ __launch_bounds__(256) void tpp_prep(
    const float* __restrict__ q, const float* __restrict__ k,
    const float* __restrict__ v, const float* __restrict__ ar,
    _Float16* __restrict__ qh, _Float16* __restrict__ kuh,
    _Float16* __restrict__ vhT) {
  const int wave = threadIdx.x >> 5;
  const int lane = threadIdx.x & 31;
  const int r    = blockIdx.x * 8 + wave;      // 0..65535
  const int bh   = r >> 11;                    // b*8+h
  const int hw   = r & 2047;
  const int b    = bh >> 3;

  const long base = (long)r * NC;
  const int c0 = lane, c1 = lane + 32;

  const float k0 = k[base + c0], k1 = k[base + c1];
  const float v0 = v[base + c0], v1 = v[base + c1];
  const float q0 = q[base + c0], q1 = q[base + c1];

  // C-dim sum of squares across the wave (32 lanes x 2 elems)
  float sk = k0 * k0 + k1 * k1;
  float sv = v0 * v0 + v1 * v1;
#pragma unroll
  for (int m = 16; m >= 1; m >>= 1) {
    sk += __shfl_xor(sk, m, 32);
    sv += __shfl_xor(sv, m, 32);
  }
  const float nk = 1.0f / (1e-6f + sqrtf(sk));
  const float nv = 1.0f / (1e-6f + sqrtf(sv));

  // H-dim norm of a_rand at (b, :, hw, c)
  float sa0 = 0.0f, sa1 = 0.0f;
#pragma unroll
  for (int hh = 0; hh < NH; ++hh) {
    const long ab = ((long)(b * NH + hh) * NHW + hw) * NC;
    const float x0 = ar[ab + c0], x1 = ar[ab + c1];
    sa0 += x0 * x0;
    sa1 += x1 * x1;
  }
  const float a0 = ar[base + c0] / (1e-6f + sqrtf(sa0));
  const float a1 = ar[base + c1] / (1e-6f + sqrtf(sa1));

  // OT-style score mod:  Km = exp(-(1-sim)/0.5) = exp(2*(sim-1))
  const float sim0 = fmaxf(k0 * nk * v0 * nv, 0.0f);
  const float sim1 = fmaxf(k1 * nk * v1 * nv, 0.0f);
  const float Km0 = __expf(2.0f * (sim0 - 1.0f));
  const float Km1 = __expf(2.0f * (sim1 - 1.0f));
  const float ku0 = k0 * k0 * v0 * a0 / (Km0 * Km0);
  const float ku1 = k1 * k1 * v1 * a1 / (Km1 * Km1);

  qh[base + c0]  = (_Float16)(q0 * 0.125f);   // fold 1/TEMPERATURE
  qh[base + c1]  = (_Float16)(q1 * 0.125f);
  kuh[base + c0] = (_Float16)ku0;
  kuh[base + c1] = (_Float16)ku1;
  // v transposed: [bh][c][hw]  -> PV B-fragments become contiguous loads
  vhT[((long)bh * NC + c0) * NHW + hw] = (_Float16)v0;
  vhT[((long)bh * NC + c1) * NHW + hw] = (_Float16)v1;
}

// ---------------------------------------------------------------------------
// Kernel 2: attention. 1 block = 128 q-rows of one (b,h); 1 wave = 16 rows.
// Phase 0: stage the full 256KB K panel into LDS with async global->LDS.
// Then 3 QK passes (max / sumexp / emit+PV) reading K from LDS only.
// Fragment layouts (M/N, C/D per ISA 7.12.2; K order contiguous, not the ISA table):
//   A 16x32 f16: lane L -> row M=L%16, halves i -> K = (L<16?0:16)+i
//                => two contiguous 8-half (16B) loads at K-offsets {base,base+8}
//   B 32x16 f16: lane L -> col N=L%16, halves i -> K = (L<16?0:16)+i (this code's order)
//                => one contiguous 16-half (32B) load
//   C/D f32 16x16: VGPR r, lane L -> (M = r + (L<16?0:8), N = L%16)
// ---------------------------------------------------------------------------
__global__ __launch_bounds__(256) void tpp_attn(
    const _Float16* __restrict__ qh, const _Float16* __restrict__ kuh,
    const _Float16* __restrict__ vhT,
    float* __restrict__ out, float* __restrict__ attn) {
  extern __shared__ char smem[];  // [0,8KB): P staging  [8KB, 264KB): K panel
  _Float16(*ldsP)[16][32] = (_Float16(*)[16][32])smem;
  _Float16* ldsK = (_Float16*)(smem + LDSP_BYTES);

  const int wave  = threadIdx.x >> 5;
  const int lane  = threadIdx.x & 31;
  const int lmod  = lane & 15;
  const bool hi   = lane >= 16;
  const int bh    = blockIdx.x >> 4;        // 0..31
  const int qblk  = blockIdx.x & 15;        // 0..15
  const int qrow0 = qblk * 128 + wave * 16; // first of this wave's 16 rows

  // ---- phase 0: async-stage K panel (2048x64 f16) into LDS ---------------
  {
    // low 32 bits of the generic shared pointer == LDS byte offset (AS3 value)
    const unsigned lds0 = (unsigned)(unsigned long long)(void*)ldsK;
    const unsigned long long g0 =
        (unsigned long long)(const void*)(kuh + (long)bh * NHW * NC);
    // 256KB in 16B chunks: 16384 chunks, 256 threads -> 64 per thread,
    // consecutive threads take consecutive chunks (512B per wave per issue).
#pragma unroll 4
    for (int i = 0; i < 64; ++i) {
      const unsigned idx = (unsigned)(i * 256 + threadIdx.x);  // chunk index
      async_g2l_b128(lds0 + idx * 16u, g0 + (unsigned long long)idx * 16u);
    }
    wait_asynccnt0();   // our own ASYNCcnt drained
    __syncthreads();    // publish LDS to all waves
  }

  // ---- A fragments of Q (already pre-scaled by 1/T) -----------------------
  const _Float16* qrow = qh + ((long)bh * NHW + qrow0 + lmod) * NC;
  const int abase = hi ? 8 : 0;   // ISA table-order K split (unused)
  const int bbase = hi ? 16 : 0;  // B-layout K split between lane halves
  union Frag { v16h f; v8h h[2]; };
  Frag aq0, aq1;
  aq0.h[0] = *(const v8h*)(qrow + bbase);
  aq0.h[1] = *(const v8h*)(qrow + bbase + 8);
  aq1.h[0] = *(const v8h*)(qrow + bbase + 32);
  aq1.h[1] = *(const v8h*)(qrow + bbase + 40);

  // ---- pass 1: per-row max over all 2048 logits ---------------------------
  float mx[8];
#pragma unroll
  for (int r = 0; r < 8; ++r) mx[r] = -3.0e38f;
  for (int j = 0; j < NHW / 16; ++j) {
    const _Float16* kcol = ldsK + (j * 16 + lmod) * NC + bbase;
    const v16h b0 = *(const v16h*)(kcol);
    const v16h b1 = *(const v16h*)(kcol + 32);
    v8f c = {};
    c = wmma_f16(aq0.f, b0, c);
    c = wmma_f16(aq1.f, b1, c);
#pragma unroll
    for (int r = 0; r < 8; ++r) mx[r] = fmaxf(mx[r], c[r]);
  }
#pragma unroll
  for (int m = 8; m >= 1; m >>= 1) {  // reduce within each 16-lane half
#pragma unroll
    for (int r = 0; r < 8; ++r) mx[r] = fmaxf(mx[r], __shfl_xor(mx[r], m, 32));
  }

  // ---- pass 2: per-row sum of exp(logit - max) ----------------------------
  float sm[8];
#pragma unroll
  for (int r = 0; r < 8; ++r) sm[r] = 0.0f;
  for (int j = 0; j < NHW / 16; ++j) {
    const _Float16* kcol = ldsK + (j * 16 + lmod) * NC + bbase;
    const v16h b0 = *(const v16h*)(kcol);
    const v16h b1 = *(const v16h*)(kcol + 32);
    v8f c = {};
    c = wmma_f16(aq0.f, b0, c);
    c = wmma_f16(aq1.f, b1, c);
#pragma unroll
    for (int r = 0; r < 8; ++r) sm[r] += __expf(c[r] - mx[r]);
  }
#pragma unroll
  for (int m = 8; m >= 1; m >>= 1) {
#pragma unroll
    for (int r = 0; r < 8; ++r) sm[r] += __shfl_xor(sm[r], m, 32);
  }
  float rinv[8];
#pragma unroll
  for (int r = 0; r < 8; ++r) rinv[r] = 1.0f / sm[r];

  // ---- pass 3: emit normalized attn + accumulate out = P @ V --------------
  v8f acc[4] = {v8f{}, v8f{}, v8f{}, v8f{}};
  const _Float16* vbase = vhT + (long)bh * NC * NHW;

  for (int jp = 0; jp < NHW / 32; ++jp) {
#pragma unroll
    for (int t = 0; t < 2; ++t) {
      const int j = jp * 2 + t;
      const _Float16* kcol = ldsK + (j * 16 + lmod) * NC + bbase;
      const v16h b0 = *(const v16h*)(kcol);
      const v16h b1 = *(const v16h*)(kcol + 32);
      v8f c = {};
      c = wmma_f16(aq0.f, b0, c);
      c = wmma_f16(aq1.f, b1, c);
#pragma unroll
      for (int r = 0; r < 8; ++r) {
        const int row = r + (hi ? 8 : 0);
        const float p = __expf(c[r] - mx[r]) * rinv[r];  // normalized prob
        // write-once 537MB stream: non-temporal (TH=NT) keeps it out of L2
        __builtin_nontemporal_store(
            p, &attn[((long)(bh * NHW + qrow0 + row)) * NHW + j * 16 + lmod]);
        ldsP[wave][row][t * 16 + lmod] = (_Float16)p;    // stage for PV
      }
    }
    // C-layout -> A-layout via per-wave LDS slice (wave-private, in-order)
    Frag ap;
    ap.h[0] = *(const v8h*)&ldsP[wave][lmod][bbase];
    ap.h[1] = *(const v8h*)&ldsP[wave][lmod][bbase + 8];
#pragma unroll
    for (int n = 0; n < 4; ++n) {  // 4 N-tiles cover C=64
      const _Float16* vc = vbase + (long)(n * 16 + lmod) * NHW + jp * 32 + bbase;
      const v16h bv = *(const v16h*)vc;
      acc[n] = wmma_f16(ap.f, bv, acc[n]);
    }
  }

  // ---- store out (C-layout f32 tiles, already normalized) -----------------
#pragma unroll
  for (int n = 0; n < 4; ++n) {
#pragma unroll
    for (int r = 0; r < 8; ++r) {
      const int row = r + (hi ? 8 : 0);
      out[((long)(bh * NHW + qrow0 + row)) * NC + n * 16 + lmod] = acc[n][r];
    }
  }
}

// ---------------------------------------------------------------------------
extern "C" void kernel_launch(void* const* d_in, const int* in_sizes, int n_in,
                              void* d_out, int out_size, void* d_ws, size_t ws_size,
                              hipStream_t stream) {
  const float* q  = (const float*)d_in[0];
  const float* k  = (const float*)d_in[1];
  const float* v  = (const float*)d_in[2];
  const float* ar = (const float*)d_in[3];

  float* out  = (float*)d_out;
  float* attn = out + NELEM;  // tuple order: (out, attn)

  _Float16* qh  = (_Float16*)d_ws;   // 8 MB
  _Float16* kuh = qh + NELEM;        // 8 MB
  _Float16* vhT = kuh + NELEM;       // 8 MB  (24 MB total workspace)

  (void)in_sizes; (void)n_in; (void)out_size; (void)ws_size;

  tpp_prep<<<NROWS / 8, 256, 0, stream>>>(q, k, v, ar, qh, kuh, vhT);
  // 264KB dynamic LDS: CDNA5 WGP allows up to 320KB per workgroup.
  tpp_attn<<<NBH * (NHW / 128), 256, SMEM_BYTES, stream>>>(qh, kuh, vhT, out, attn);
}
